// TacotronDecoder_85177791414758
// MI455X (gfx1250) — matrix operand order repaired and checked by reading, not compiled, not run on hardware
//
#include <hip/hip_runtime.h>
#include <hip/hip_bf16.h>

// ---------------- problem constants ----------------
#define B64   64
#define TIN   200
#define TOUT  400
#define DENC  512
#define DRNN  1024
#define PRE   256
#define ATT   128
#define NFILT 32
#define KS    31
#define NMEL  80
#define KA    1792   // PRE + DENC + DRNN   (attention-LSTM concat K)
#define KD    2560   // DRNN + DENC + DRNN  (decoder-LSTM concat K)
#define NG    4096   // 4*DRNN gate width
#define NBLK  64
#define NTHR  256
#define NTOT  (NBLK*NTHR)

typedef __bf16 bf16_t;
typedef __bf16 v16bf __attribute__((ext_vector_type(16)));
typedef __bf16 v8bf  __attribute__((ext_vector_type(8)));
typedef float  v8f   __attribute__((ext_vector_type(8)));

__device__ __forceinline__ float sigm(float x) { return 1.0f / (1.0f + expf(-x)); }

// ---------------- grid-wide barrier (persistent kernel) ----------------
__device__ __forceinline__ void gbar(unsigned* cnt, unsigned* epoch) {
  __syncthreads();
  if (threadIdx.x == 0) {
    __threadfence();                       // release
    unsigned e = ++(*epoch);
    atomicAdd(cnt, 1u);
    const unsigned target = e * (unsigned)NBLK;
    while (*(volatile unsigned*)cnt < target) __builtin_amdgcn_s_sleep(1);
    __threadfence();                       // acquire
  }
  __syncthreads();
}

// ---------------- WMMA helpers (A/B: contiguous k order used by this code) ----------------
// A 16x32 bf16 frag: M=lane&15, 16 contiguous K starting at (lane>>4)*16 (same order as B)
// B 32x16 bf16 frag: N=lane&15, 16 contiguous K starting at (lane>>4)*16
// D 16x16 f32: VGPR r -> M=(lane>>4)*8+r, N=lane&15

// each wave computes two 16x16 tiles (same N-tile, two M-tiles) of G(64 x NG) = X(64 x K) * W(NG x K)^T
__device__ __forceinline__ void gemm_dual(const bf16_t* __restrict__ X,
                                          const bf16_t* __restrict__ W,
                                          float* __restrict__ G,
                                          int K, int wave, int lane) {
  const int tt0  = wave * 2;                 // tile ids tt0, tt0+1 (1024 tiles, 512 waves)
  const int m0   = (tt0 & 3) * 16;
  const int m1   = ((tt0 + 1) & 3) * 16;
  const int n0   = (tt0 >> 2) * 16;          // same for both tiles
  const int row  = lane & 15;
  const int half = lane >> 4;
  const bf16_t* pB  = W + (size_t)(n0 + row) * K + half * 16;
  const bf16_t* pA0 = X + (size_t)(m0 + row) * K + half * 16;
  const bf16_t* pA1 = X + (size_t)(m1 + row) * K + half * 16;
  v8f c0 = {0.f,0.f,0.f,0.f,0.f,0.f,0.f,0.f};
  v8f c1 = {0.f,0.f,0.f,0.f,0.f,0.f,0.f,0.f};
  for (int k0 = 0; k0 < K; k0 += 32) {
    v16bf b = *(const v16bf*)(pB + k0);
    __builtin_prefetch(pB + k0 + 32, 0, 0);        // global_prefetch_b8
    v8bf lo0 = *(const v8bf*)(pA0 + k0);
    v8bf hi0 = *(const v8bf*)(pA0 + k0 + 8);
    v8bf lo1 = *(const v8bf*)(pA1 + k0);
    v8bf hi1 = *(const v8bf*)(pA1 + k0 + 8);
    v16bf a0, a1;
#pragma unroll
    for (int i = 0; i < 8; ++i) { a0[i] = lo0[i]; a0[i+8] = hi0[i];
                                  a1[i] = lo1[i]; a1[i+8] = hi1[i]; }
    c0 = __builtin_amdgcn_wmma_f32_16x16x32_bf16(false, a0, false, b, (short)0, c0, false, false);
    c1 = __builtin_amdgcn_wmma_f32_16x16x32_bf16(false, a1, false, b, (short)0, c1, false, false);
  }
  const int col = lane & 15;
  const int rb  = half * 8;
#pragma unroll
  for (int r = 0; r < 8; ++r) {
    G[(size_t)(m0 + rb + r) * NG + n0 + col] = c0[r];
    G[(size_t)(m1 + rb + r) * NG + n0 + col] = c1[r];
  }
}

// ---------------- precompute kernels ----------------
__global__ __launch_bounds__(256) void cvt_f32_bf16(const float* __restrict__ s,
                                                    bf16_t* __restrict__ d, int n) {
  int i = blockIdx.x * 256 + threadIdx.x;
  if (i < n) d[i] = (bf16_t)s[i];
}

// dst (rows x (ka+kb)) = [A | B] converted to bf16 (weight concat for fused ih|hh GEMM)
__global__ __launch_bounds__(256) void cvt_cat2(const float* __restrict__ A, int ka,
                                                const float* __restrict__ Bm, int kb,
                                                bf16_t* __restrict__ dst, int rows) {
  int i = blockIdx.x * 256 + threadIdx.x;
  int K = ka + kb;
  if (i >= rows * K) return;
  int r = i / K, k = i - r * K;
  float v = (k < ka) ? A[(size_t)r * ka + k] : Bm[(size_t)r * kb + (k - ka)];
  dst[i] = (bf16_t)v;
}

// prenet for all 400 steps: xs[t,b,:] = relu(relu(di @ Wp1^T) @ Wp2^T), di[0]=0, di[t]=dec_in[t-1]
__global__ __launch_bounds__(256) void prenet_kernel(const float* __restrict__ dec_in,
                                                     const float* __restrict__ Wp1,
                                                     const float* __restrict__ Wp2,
                                                     bf16_t* __restrict__ xs) {
  __shared__ float din[NMEL];
  __shared__ float h1[PRE];
  const int r = blockIdx.x;          // r = t*64 + b
  const int t = r >> 6, b = r & 63;
  const int tid = threadIdx.x;
  if (tid < NMEL) din[tid] = (t == 0) ? 0.f : dec_in[((size_t)(t - 1) * B64 + b) * NMEL + tid];
  __syncthreads();
  float acc = 0.f;
  const float* w1 = Wp1 + (size_t)tid * NMEL;
#pragma unroll
  for (int k = 0; k < NMEL; ++k) acc += din[k] * w1[k];
  h1[tid] = fmaxf(acc, 0.f);
  __syncthreads();
  acc = 0.f;
  const float* w2 = Wp2 + (size_t)tid * PRE;
  for (int k = 0; k < PRE; ++k) acc += h1[k] * w2[k];
  xs[(size_t)r * PRE + tid] = (bf16_t)fmaxf(acc, 0.f);
}

// processed_memory: pm(12800 x 128) = enc_bf(12800 x 512) @ Wmem_bf(128 x 512)^T  via WMMA
__global__ __launch_bounds__(256) void pm_kernel(const bf16_t* __restrict__ enc,
                                                 const bf16_t* __restrict__ wm,
                                                 float* __restrict__ pm) {
  const int lane = threadIdx.x & 31;
  const int wave = threadIdx.x >> 5;         // 8 waves -> N tiles 0..7 (ATT=128)
  const int m0 = blockIdx.x * 16;
  const int n0 = wave * 16;
  const int row = lane & 15, half = lane >> 4;
  const bf16_t* pA = enc + (size_t)(m0 + row) * DENC + half * 16;
  const bf16_t* pB = wm  + (size_t)(n0 + row) * DENC + half * 16;
  v8f c = {0.f,0.f,0.f,0.f,0.f,0.f,0.f,0.f};
  for (int k0 = 0; k0 < DENC; k0 += 32) {
    v16bf b = *(const v16bf*)(pB + k0);
    v8bf lo = *(const v8bf*)(pA + k0);
    v8bf hi = *(const v8bf*)(pA + k0 + 8);
    v16bf a;
#pragma unroll
    for (int i = 0; i < 8; ++i) { a[i] = lo[i]; a[i+8] = hi[i]; }
    c = __builtin_amdgcn_wmma_f32_16x16x32_bf16(false, a, false, b, (short)0, c, false, false);
  }
  const int col = lane & 15, rb = half * 8;
#pragma unroll
  for (int r = 0; r < 8; ++r)
    pm[(size_t)(m0 + rb + r) * ATT + n0 + col] = c[r];
}

// ---------------- persistent sequential decoder ----------------
struct DecArgs {
  const bf16_t* enc_bf; const int* lens;
  const float *b_a, *Wq, *v, *Wconv, *Wloc, *b_d, *Wp, *bp;
  const bf16_t *xs, *Wa, *Wd; const float* pm;
  float *ah, *ac, *dh, *dc, *actx, *aw, *awcum;
  float *g_a, *g_d, *q, *e;
  bf16_t *xa, *xd;
  unsigned* cnt;
  float *out_mel, *out_align;
};

__global__ __launch_bounds__(NTHR) void decoder_kernel(DecArgs A) {
  __shared__ float sWloc[ATT * NFILT];
  __shared__ float sWconv[NFILT * 62];
  __shared__ float sV[ATT];
  __shared__ float sRed[NTHR];
  __shared__ float sAw[256];
  __shared__ unsigned sEpoch;

  const int tid  = threadIdx.x;
  const int tidg = blockIdx.x * NTHR + tid;
  const int lane = tid & 31;
  const int wave = blockIdx.x * (NTHR / 32) + (tid >> 5);

  if (tid == 0) sEpoch = 0;
  for (int i = tid; i < ATT * NFILT; i += NTHR) sWloc[i]  = A.Wloc[i];
  for (int i = tid; i < NFILT * 62;  i += NTHR) sWconv[i] = A.Wconv[i];
  for (int i = tid; i < ATT;         i += NTHR) sV[i]     = A.v[i];
  __syncthreads();

  for (int t = 0; t < TOUT; ++t) {
    // ---- Phase 1: stage xa = [xs_t | actx | ah] as bf16 (64 x KA)
    {
      const bf16_t* xsrow = A.xs + (size_t)t * B64 * PRE;
      for (int i = tidg; i < B64 * KA; i += NTOT) {
        int b = i / KA, c = i - b * KA;
        bf16_t vb;
        if (c < PRE)              vb = xsrow[b * PRE + c];
        else if (c < PRE + DENC)  vb = (bf16_t)A.actx[b * DENC + (c - PRE)];
        else                      vb = (bf16_t)A.ah[b * DRNN + (c - PRE - DENC)];
        A.xa[i] = vb;
      }
    }
    gbar(A.cnt, &sEpoch);

    // ---- Phase 2: attention-LSTM gates g_a = xa @ [Wih|Whh]^T (WMMA)
    gemm_dual(A.xa, A.Wa, A.g_a, KA, wave, lane);
    gbar(A.cnt, &sEpoch);

    // ---- Phase 3: LSTM-A elementwise
    for (int i = tidg; i < B64 * DRNN; i += NTOT) {
      int b = i >> 10, j = i & 1023;
      const float* g = A.g_a + (size_t)b * NG;
      float gi = g[j]          + A.b_a[j];
      float gf = g[1024 + j]   + A.b_a[1024 + j];
      float gg = g[2048 + j]   + A.b_a[2048 + j];
      float go = g[3072 + j]   + A.b_a[3072 + j];
      float c  = sigm(gf) * A.ac[i] + sigm(gi) * tanhf(gg);
      A.ac[i] = c;
      A.ah[i] = sigm(go) * tanhf(c);
    }
    gbar(A.cnt, &sEpoch);

    // ---- Phase 4a: q = ah @ Wq^T (64 x 128)
    for (int i = tidg; i < B64 * ATT; i += NTOT) {
      int b = i >> 7, d = i & 127;
      const float* h = A.ah + (size_t)b * DRNN;
      const float* w = A.Wq + (size_t)d * DRNN;
      float acc = 0.f;
      for (int k = 0; k < DRNN; ++k) acc += h[k] * w[k];
      A.q[i] = acc;
    }
    gbar(A.cnt, &sEpoch);

    // ---- Phase 4b: energies e[b,t'] = v . tanh(q + pm + conv([aw;awcum]) @ Wloc^T)
    for (int p = tidg; p < B64 * TIN; p += NTOT) {
      int b = p / TIN, tt = p - b * TIN;
      float conv[NFILT];
#pragma unroll
      for (int f = 0; f < NFILT; ++f) conv[f] = 0.f;
      const float* awb = A.aw    + (size_t)b * TIN;
      const float* awc = A.awcum + (size_t)b * TIN;
      for (int k = 0; k < KS; ++k) {
        int s = tt + k - 15;
        if (s >= 0 && s < TIN) {
          float a0 = awb[s], a1 = awc[s];
#pragma unroll
          for (int f = 0; f < NFILT; ++f)
            conv[f] += a0 * sWconv[f * 62 + k] + a1 * sWconv[f * 62 + 31 + k];
        }
      }
      const float* qb  = A.q  + (size_t)b * ATT;
      const float* pmb = A.pm + (size_t)p * ATT;
      float eacc = 0.f;
      for (int d = 0; d < ATT; ++d) {
        const float* wl = sWloc + d * NFILT;
        float lp = 0.f;
#pragma unroll
        for (int f = 0; f < NFILT; ++f) lp += conv[f] * wl[f];
        eacc += sV[d] * tanhf(qb[d] + pmb[d] + lp);
      }
      A.e[p] = eacc;
    }
    gbar(A.cnt, &sEpoch);

    // ---- Phase 5: masked softmax + context + cumulative attention (block b)
    {
      const int b = blockIdx.x;
      const int len = A.lens[b];
      float ev = -3.4e38f;
      if (tid < TIN) ev = (tid < len) ? A.e[b * TIN + tid] : -1e8f;
      sRed[tid] = ev; __syncthreads();
      for (int s = 128; s > 0; s >>= 1) { if (tid < s) sRed[tid] = fmaxf(sRed[tid], sRed[tid + s]); __syncthreads(); }
      const float m = sRed[0]; __syncthreads();
      float ex = (tid < TIN) ? expf(ev - m) : 0.f;
      sRed[tid] = ex; __syncthreads();
      for (int s = 128; s > 0; s >>= 1) { if (tid < s) sRed[tid] += sRed[tid + s]; __syncthreads(); }
      const float inv = 1.f / sRed[0]; __syncthreads();
      if (tid < TIN) {
        float awv = ex * inv;
        sAw[tid] = awv;
        A.aw[b * TIN + tid] = awv;
        A.awcum[b * TIN + tid] += awv;
        A.out_align[((size_t)b * TOUT + t) * TIN + tid] = awv;
      }
      __syncthreads();
      for (int e0 = tid; e0 < DENC; e0 += NTHR) {
        const bf16_t* pe = A.enc_bf + (size_t)b * TIN * DENC + e0;
        float acc = 0.f;
        for (int ti = 0; ti < TIN; ++ti) acc += sAw[ti] * (float)pe[(size_t)ti * DENC];
        A.actx[b * DENC + e0] = acc;
      }
    }
    gbar(A.cnt, &sEpoch);

    // ---- Phase 6: stage xd = [ah | actx | dh] bf16 (64 x KD)
    for (int i = tidg; i < B64 * KD; i += NTOT) {
      int b = i / KD, c = i - b * KD;
      float vv;
      if (c < DRNN)              vv = A.ah[b * DRNN + c];
      else if (c < DRNN + DENC)  vv = A.actx[b * DENC + (c - DRNN)];
      else                       vv = A.dh[b * DRNN + (c - DRNN - DENC)];
      A.xd[i] = (bf16_t)vv;
    }
    gbar(A.cnt, &sEpoch);

    // ---- Phase 7: decoder-LSTM gates (WMMA)
    gemm_dual(A.xd, A.Wd, A.g_d, KD, wave, lane);
    gbar(A.cnt, &sEpoch);

    // ---- Phase 8: LSTM-D elementwise
    for (int i = tidg; i < B64 * DRNN; i += NTOT) {
      int b = i >> 10, j = i & 1023;
      const float* g = A.g_d + (size_t)b * NG;
      float gi = g[j]          + A.b_d[j];
      float gf = g[1024 + j]   + A.b_d[1024 + j];
      float gg = g[2048 + j]   + A.b_d[2048 + j];
      float go = g[3072 + j]   + A.b_d[3072 + j];
      float c  = sigm(gf) * A.dc[i] + sigm(gi) * tanhf(gg);
      A.dc[i] = c;
      A.dh[i] = sigm(go) * tanhf(c);
    }
    gbar(A.cnt, &sEpoch);

    // ---- Phase 9: projection out = [dh|actx] @ Wp^T + bp
    for (int i = tidg; i < B64 * NMEL; i += NTOT) {
      int b = i / NMEL, m = i - b * NMEL;
      const float* w  = A.Wp + (size_t)m * (DRNN + DENC);
      const float* h  = A.dh   + (size_t)b * DRNN;
      const float* cx = A.actx + (size_t)b * DENC;
      float acc = A.bp[m];
      for (int k = 0; k < DRNN; ++k) acc += h[k] * w[k];
      for (int k = 0; k < DENC; ++k) acc += cx[k] * w[DRNN + k];
      A.out_mel[((size_t)b * TOUT + t) * NMEL + m] = acc;
    }
    gbar(A.cnt, &sEpoch);
  }
}

// ---------------- host launch ----------------
extern "C" void kernel_launch(void* const* d_in, const int* in_sizes, int n_in,
                              void* d_out, int out_size, void* d_ws, size_t ws_size,
                              hipStream_t stream) {
  (void)in_sizes; (void)n_in; (void)out_size; (void)ws_size;
  const float* enc    = (const float*)d_in[0];
  const float* decin  = (const float*)d_in[1];
  const int*   lens   = (const int*)  d_in[2];
  const float* W_p1   = (const float*)d_in[3];
  const float* W_p2   = (const float*)d_in[4];
  const float* W_ih_a = (const float*)d_in[5];
  const float* W_hh_a = (const float*)d_in[6];
  const float* b_a    = (const float*)d_in[7];
  const float* Wq     = (const float*)d_in[8];
  const float* Wmem   = (const float*)d_in[9];
  const float* vvec   = (const float*)d_in[10];
  const float* Wconv  = (const float*)d_in[11];
  const float* Wloc   = (const float*)d_in[12];
  const float* W_ih_d = (const float*)d_in[13];
  const float* W_hh_d = (const float*)d_in[14];
  const float* b_d    = (const float*)d_in[15];
  const float* Wp     = (const float*)d_in[16];
  const float* bp     = (const float*)d_in[17];

  char* ws = (char*)d_ws;
  size_t off = 0;
  auto take = [&](size_t bytes) -> char* {
    char* p = ws + off;
    off += (bytes + 255) & ~(size_t)255;
    return p;
  };

  // --- state region first (single memset zero-init, graph-capture safe) ---
  unsigned* cnt   = (unsigned*)take(256);
  float* ah    = (float*)take((size_t)B64 * DRNN * 4);
  float* ac    = (float*)take((size_t)B64 * DRNN * 4);
  float* dh    = (float*)take((size_t)B64 * DRNN * 4);
  float* dc    = (float*)take((size_t)B64 * DRNN * 4);
  float* actx  = (float*)take((size_t)B64 * DENC * 4);
  float* aw    = (float*)take((size_t)B64 * TIN  * 4);
  float* awcum = (float*)take((size_t)B64 * TIN  * 4);
  const size_t state_bytes = off;

  float*  g_a  = (float*)take((size_t)B64 * NG * 4);
  float*  g_d  = (float*)take((size_t)B64 * NG * 4);
  float*  qb   = (float*)take((size_t)B64 * ATT * 4);
  float*  eb   = (float*)take((size_t)B64 * TIN * 4);
  bf16_t* xa   = (bf16_t*)take((size_t)B64 * KA * 2);
  bf16_t* xd   = (bf16_t*)take((size_t)B64 * KD * 2);
  bf16_t* xs   = (bf16_t*)take((size_t)TOUT * B64 * PRE * 2);
  bf16_t* Wa   = (bf16_t*)take((size_t)NG * KA * 2);
  bf16_t* Wd   = (bf16_t*)take((size_t)NG * KD * 2);
  bf16_t* encb = (bf16_t*)take((size_t)B64 * TIN * DENC * 2);
  bf16_t* wmb  = (bf16_t*)take((size_t)ATT * DENC * 2);
  float*  pmb  = (float*)take((size_t)B64 * TIN * ATT * 4);

  hipMemsetAsync(d_ws, 0, state_bytes, stream);

  // weight / activation precompute
  cvt_cat2<<<((size_t)NG * KA + 255) / 256, 256, 0, stream>>>(W_ih_a, 768,  W_hh_a, 1024, Wa, NG);
  cvt_cat2<<<((size_t)NG * KD + 255) / 256, 256, 0, stream>>>(W_ih_d, 1536, W_hh_d, 1024, Wd, NG);
  cvt_f32_bf16<<<((size_t)B64 * TIN * DENC + 255) / 256, 256, 0, stream>>>(enc, encb, B64 * TIN * DENC);
  cvt_f32_bf16<<<((size_t)ATT * DENC + 255) / 256, 256, 0, stream>>>(Wmem, wmb, ATT * DENC);
  prenet_kernel<<<TOUT * B64, 256, 0, stream>>>(decin, W_p1, W_p2, xs);
  pm_kernel<<<(B64 * TIN) / 16, 256, 0, stream>>>(encb, wmb, pmb);

  // persistent sequential decoder
  DecArgs A;
  A.enc_bf = encb; A.lens = lens;
  A.b_a = b_a; A.Wq = Wq; A.v = vvec; A.Wconv = Wconv; A.Wloc = Wloc;
  A.b_d = b_d; A.Wp = Wp; A.bp = bp;
  A.xs = xs; A.Wa = Wa; A.Wd = Wd; A.pm = pmb;
  A.ah = ah; A.ac = ac; A.dh = dh; A.dc = dc; A.actx = actx; A.aw = aw; A.awcum = awcum;
  A.g_a = g_a; A.g_d = g_d; A.q = qb; A.e = eb;
  A.xa = xa; A.xd = xd; A.cnt = cnt;
  A.out_mel = (float*)d_out;
  A.out_align = (float*)d_out + (size_t)B64 * TOUT * NMEL;

  decoder_kernel<<<NBLK, NTHR, 0, stream>>>(A);
}
